// MultiHeadSelfAttention_70351564308780
// MI455X (gfx1250) — matrix operand order repaired and checked by reading, not compiled, not run on hardware
//
#include <hip/hip_runtime.h>

// MI455X / gfx1250 multi-head self-attention.
// f32->f16 convert -> QKV WMMA GEMMs -> flash-attention (WMMA) -> out-proj WMMA GEMM.
// Matrix math: v_wmma_f32_16x16x32_f16.  Shared operands staged to LDS with
// double-buffered global_load_async_to_lds_b128 (ASYNCcnt) so copies overlap WMMA.
// B/K/V fragments are loaded in batches so DS loads clause together and WMMAs
// issue back-to-back under a single s_wait_dscnt.

typedef __attribute__((ext_vector_type(16))) _Float16 v16h;
typedef __attribute__((ext_vector_type(8)))  _Float16 v8h;
typedef __attribute__((ext_vector_type(8)))  float    v8f;

#define FEATS 1024
#define HEADS 16
#define DHEAD 64
#define BATCH 2
#define SEQ   2048
#define NTOK  (BATCH * SEQ)   // 4096 rows

// ---- fragment helpers (layouts per CDNA5 ISA 7.12.2) -----------------------

__device__ __forceinline__ v16h frag_cat(v8h lo, v8h hi) {
  return __builtin_shufflevector(lo, hi, 0,1,2,3,4,5,6,7,8,9,10,11,12,13,14,15);
}

// A fragment: 16x32 f16, row-major, stride lda (halfs).
// M = lane%16; lanes 0-15 hold K {0..7,16..23}, lanes 16-31 hold K {8..15,24..31}.
__device__ __forceinline__ v16h load_a(const _Float16* base, int lda) {
  const int lane = threadIdx.x & 31;
  const int r = lane & 15, hb = lane >> 4;
  const _Float16* p = base + (size_t)r * lda + hb * 8;
  v8h lo = *(const v8h*)(p);
  v8h hi = *(const v8h*)(p + 16);
  return frag_cat(lo, hi);
}

// B fragment: 32x16 f16; element (K,N) read from base[n*ldb + k].
// N = lane%16; lanes 0-15 hold K {0..7,16..23}, lanes 16-31 hold K {8..15,24..31}.
// Works for both global and LDS source pointers.
__device__ __forceinline__ v16h load_b(const _Float16* base, int ldb) {
  const int lane = threadIdx.x & 31;
  const int n = lane & 15, hb = lane >> 4;
  const _Float16* p = base + (size_t)n * ldb + hb * 8;
  v8h lo = *(const v8h*)(p);
  v8h hi = *(const v8h*)(p + 16);
  return frag_cat(lo, hi);
}

__device__ __forceinline__ v8f wmma_f16(v16h a, v16h b, v8f c) {
  return __builtin_amdgcn_wmma_f32_16x16x32_f16(false, a, false, b, (short)0, c, false, false);
}

// ---- gfx1250 async global->LDS copy (ASYNCcnt) -----------------------------

__device__ __forceinline__ unsigned lds_off(const void* p) {
  // flat LDS-aperture address: low 32 bits are the LDS byte offset
  return (unsigned)(size_t)p;
}
__device__ __forceinline__ void async_b128(unsigned dst_lds, const _Float16* src) {
  asm volatile("global_load_async_to_lds_b128 %0, %1, off"
               :: "v"(dst_lds), "v"(src) : "memory");
}
__device__ __forceinline__ void wait_async0() {
  asm volatile("s_wait_asynccnt 0x0" ::: "memory");
}

// ---- kernel 0: f32 -> f16 convert ------------------------------------------

__global__ void cvt_f32_f16(const float* __restrict__ s, _Float16* __restrict__ d, int n) {
  int i = blockIdx.x * blockDim.x + threadIdx.x;
  const int stride = gridDim.x * blockDim.x;
  for (; i < n; i += stride) d[i] = (_Float16)s[i];
}

// ---- kernel 1: QKV projection ----------------------------------------------
// Block: 8 waves share one 64-col group of W (staged in LDS, async double-buffered);
// each wave owns one 16-row tile.  which: 0/1 -> [B,H,N,DH]; 2 -> V^T [B,H,DH,N].

#define BPAD 40   // padded row stride (halfs) for 32-half rows: conflict-free b128 reads

__global__ void __launch_bounds__(256) qkv_gemm(const _Float16* __restrict__ xh,
                                                const _Float16* __restrict__ Wh,
                                                const float* __restrict__ bias,
                                                _Float16* __restrict__ dst, int which) {
  __shared__ alignas(16) _Float16 bbuf[2][64 * BPAD];
  const int tid = threadIdx.x;
  const int cg = blockIdx.x & 15;                       // 64-col group
  const int rowtile = (blockIdx.x >> 4) * 8 + (tid >> 5);

  const _Float16* Wbase = Wh + (size_t)cg * 64 * FEATS;
  // stage: 64 rows x 32 halfs = 256 x 16B chunks, one per thread
  const int brow = tid >> 2, bch = tid & 3;
  const _Float16* bsrc = Wbase + (size_t)brow * FEATS + bch * 8;
  const unsigned bdst0 = lds_off(&bbuf[0][brow * BPAD + bch * 8]);
  const unsigned bdst1 = lds_off(&bbuf[1][brow * BPAD + bch * 8]);

  async_b128(bdst0, bsrc);                              // kb = 0

  const _Float16* Abase = xh + (size_t)rowtile * 16 * FEATS;
  v8f acc[4] = {};
  v16h a_cur = load_a(Abase, FEATS);
  int cur = 0;
  for (int kb = 0; kb < FEATS; kb += 32) {
    wait_async0();
    __syncthreads();                                    // bbuf[cur] ready; prev reads done
    if (kb + 32 < FEATS)
      async_b128(cur ? bdst0 : bdst1, bsrc + kb + 32);
    v16h a_next = a_cur;
    if (kb + 32 < FEATS) a_next = load_a(Abase + kb + 32, FEATS);
    const _Float16* bb = &bbuf[cur][0];
    v16h bf[4];
#pragma unroll
    for (int t = 0; t < 4; ++t)                         // batch DS loads (one clause)
      bf[t] = load_b(bb + t * 16 * BPAD, BPAD);
#pragma unroll
    for (int t = 0; t < 4; ++t)                         // back-to-back WMMAs
      acc[t] = wmma_f16(a_cur, bf[t], acc[t]);
    a_cur = a_next;
    cur ^= 1;
  }

  const int lane = tid & 31;
  const int nloc = lane & 15, hb = lane >> 4;
#pragma unroll
  for (int t = 0; t < 4; ++t) {
    const int col = cg * 64 + t * 16 + nloc;
    const float bc = bias[col];
    const int h = col >> 6, d = col & 63;
#pragma unroll
    for (int v = 0; v < 8; ++v) {
      const int row = rowtile * 16 + v + 8 * hb;        // token = b*SEQ + i
      const int b = row >> 11, i = row & 2047;
      const float val = acc[t][v] + bc;
      if (which == 2)
        dst[((size_t)((b * HEADS + h) * DHEAD + d)) * SEQ + i] = (_Float16)val;  // V^T
      else
        dst[((size_t)((b * HEADS + h) * SEQ) + i) * DHEAD + d] = (_Float16)val;  // Q/K
    }
  }
}

// ---- kernel 2: flash attention ---------------------------------------------
// 8 waves/block share (b,h): K/V tiles for each 32-key step staged in LDS
// (async, double-buffered); each wave owns a 16-row query tile.

#define KPAD 72   // padded stride for 64-half K rows
#define VPAD 40   // padded stride for 32-half V^T rows

__global__ void __launch_bounds__(256) flash_attn(const _Float16* __restrict__ Q,   // [B,H,N,DH]
                                                  const _Float16* __restrict__ K,   // [B,H,N,DH]
                                                  const _Float16* __restrict__ Vt,  // [B,H,DH,N]
                                                  _Float16* __restrict__ AO) {      // [B,N,F]
  __shared__ alignas(16) _Float16 kbuf[2][32 * KPAD];
  __shared__ alignas(16) _Float16 vbuf[2][64 * VPAD];
  __shared__ alignas(16) _Float16 pbuf[8][16 * 40];
  const int tid = threadIdx.x;
  const int wave = tid >> 5;
  const int bh = blockIdx.x >> 4;                       // all 8 waves share (b,h)
  const int qt = (blockIdx.x & 15) * 8 + wave;          // query tile 0..127
  const int b = bh >> 4, h = bh & 15;
  const _Float16* Qb = Q + (size_t)bh * SEQ * DHEAD + (size_t)qt * 16 * DHEAD;
  const _Float16* Kb = K + (size_t)bh * SEQ * DHEAD;
  const _Float16* Vb = Vt + (size_t)bh * DHEAD * SEQ;
  const int lane = tid & 31;
  const int nloc = lane & 15, hb = lane >> 4;

  // stage indices: K tile 32x64 halfs = 256 chunks; V tile 64x32 halfs = 256 chunks
  const int krow = tid >> 3, kch = tid & 7;
  const int vrow = tid >> 2, vch = tid & 3;
  const _Float16* ksrc = Kb + (size_t)krow * DHEAD + kch * 8;
  const _Float16* vsrc = Vb + (size_t)vrow * SEQ + vch * 8;
  const unsigned kdst[2] = { lds_off(&kbuf[0][krow * KPAD + kch * 8]),
                             lds_off(&kbuf[1][krow * KPAD + kch * 8]) };
  const unsigned vdst[2] = { lds_off(&vbuf[0][vrow * VPAD + vch * 8]),
                             lds_off(&vbuf[1][vrow * VPAD + vch * 8]) };

  async_b128(kdst[0], ksrc);                            // j0 = 0
  async_b128(vdst[0], vsrc);

  const v16h qf0 = load_a(Qb + 0,  DHEAD);              // d = 0..31
  const v16h qf1 = load_a(Qb + 32, DHEAD);              // d = 32..63

  v8f acc0 = {}, acc1 = {}, acc2 = {}, acc3 = {};
  float m[8], l[8];
#pragma unroll
  for (int v = 0; v < 8; ++v) { m[v] = -1e30f; l[v] = 0.0f; }
  const float scale = 1.0f / 32.0f;                     // 1/sqrt(FEATS)

  _Float16* myl = pbuf[wave];
  int cur = 0;
  for (int j0 = 0; j0 < SEQ; j0 += 32) {
    wait_async0();
    __syncthreads();                                    // K/V buf[cur] ready
    if (j0 + 32 < SEQ) {
      async_b128(kdst[cur ^ 1], ksrc + (size_t)32 * DHEAD + j0 * DHEAD);
      async_b128(vdst[cur ^ 1], vsrc + j0 + 32);
    }

    // S = Q K^T for 32 keys from LDS (two 16x16 C-tiles); batch the 4 K-frags
    const _Float16* kb0 = &kbuf[cur][0];
    v16h kf[4];
    kf[0] = load_b(kb0 + 0,              KPAD);         // rows j0..+15,  d 0..31
    kf[1] = load_b(kb0 + 32,             KPAD);         // rows j0..+15,  d 32..63
    kf[2] = load_b(kb0 + 16 * KPAD + 0,  KPAD);         // rows j0+16..+31, d 0..31
    kf[3] = load_b(kb0 + 16 * KPAD + 32, KPAD);         // rows j0+16..+31, d 32..63
    v8f s0 = {}, s1 = {};
    s0 = wmma_f16(qf0, kf[0], s0);
    s0 = wmma_f16(qf1, kf[1], s0);
    s1 = wmma_f16(qf0, kf[2], s1);
    s1 = wmma_f16(qf1, kf[3], s1);

    // online softmax: per C-row (row = v + 8*hb) reduce across the 16-lane group
    float r[8];
#pragma unroll
    for (int v = 0; v < 8; ++v) {
      const float a0 = s0[v] * scale, a1 = s1[v] * scale;
      float mx = fmaxf(a0, a1);
      mx = fmaxf(mx, __shfl_xor(mx, 1, 32));
      mx = fmaxf(mx, __shfl_xor(mx, 2, 32));
      mx = fmaxf(mx, __shfl_xor(mx, 4, 32));
      mx = fmaxf(mx, __shfl_xor(mx, 8, 32));
      const float mn = fmaxf(m[v], mx);
      r[v] = __expf(m[v] - mn);
      const float p0 = __expf(a0 - mn), p1 = __expf(a1 - mn);
      float rs = p0 + p1;
      rs += __shfl_xor(rs, 1, 32);
      rs += __shfl_xor(rs, 2, 32);
      rs += __shfl_xor(rs, 4, 32);
      rs += __shfl_xor(rs, 8, 32);
      l[v] = l[v] * r[v] + rs;
      m[v] = mn;
      myl[(v + 8 * hb) * 40 + nloc]      = (_Float16)p0;   // P in C-layout -> LDS
      myl[(v + 8 * hb) * 40 + nloc + 16] = (_Float16)p1;
    }
#pragma unroll
    for (int v = 0; v < 8; ++v) {
      acc0[v] *= r[v]; acc1[v] *= r[v]; acc2[v] *= r[v]; acc3[v] *= r[v];
    }

    asm volatile("s_wait_dscnt 0x0" ::: "memory");      // per-wave P store->load handoff

    // P as A fragment (16x32, lda = 40 halfs); batch the 4 V-frags
    const _Float16* pp = myl + (size_t)(lane & 15) * 40 + (hb ? 8 : 0);
    const v16h pf = frag_cat(*(const v8h*)pp, *(const v8h*)(pp + 16));

    const _Float16* vb0 = &vbuf[cur][0];
    v16h vf[4];
#pragma unroll
    for (int t = 0; t < 4; ++t)
      vf[t] = load_b(vb0 + t * 16 * VPAD, VPAD);
    acc0 = wmma_f16(pf, vf[0], acc0);
    acc1 = wmma_f16(pf, vf[1], acc1);
    acc2 = wmma_f16(pf, vf[2], acc2);
    acc3 = wmma_f16(pf, vf[3], acc3);
    cur ^= 1;
  }

  // write attn output [b, i, h*64 + d] as f16 for the final projection
#pragma unroll
  for (int v = 0; v < 8; ++v) {
    const int i = qt * 16 + v + 8 * hb;
    const float inv = 1.0f / l[v];
    const size_t base = ((size_t)(b * SEQ + i)) * FEATS + h * DHEAD + nloc;
    AO[base + 0]  = (_Float16)(acc0[v] * inv);
    AO[base + 16] = (_Float16)(acc1[v] * inv);
    AO[base + 32] = (_Float16)(acc2[v] * inv);
    AO[base + 48] = (_Float16)(acc3[v] * inv);
  }
}

// ---- kernel 3: output projection (fp32 out) --------------------------------

__global__ void __launch_bounds__(256) out_gemm(const _Float16* __restrict__ AO,
                                                const _Float16* __restrict__ Woh,
                                                const float* __restrict__ bo,
                                                float* __restrict__ out) {
  __shared__ alignas(16) _Float16 bbuf[2][64 * BPAD];
  const int tid = threadIdx.x;
  const int cg = blockIdx.x & 15;
  const int rowtile = (blockIdx.x >> 4) * 8 + (tid >> 5);

  const _Float16* Wbase = Woh + (size_t)cg * 64 * FEATS;
  const int brow = tid >> 2, bch = tid & 3;
  const _Float16* bsrc = Wbase + (size_t)brow * FEATS + bch * 8;
  const unsigned bdst0 = lds_off(&bbuf[0][brow * BPAD + bch * 8]);
  const unsigned bdst1 = lds_off(&bbuf[1][brow * BPAD + bch * 8]);

  async_b128(bdst0, bsrc);

  const _Float16* Abase = AO + (size_t)rowtile * 16 * FEATS;
  v8f acc[4] = {};
  v16h a_cur = load_a(Abase, FEATS);
  int cur = 0;
  for (int kb = 0; kb < FEATS; kb += 32) {
    wait_async0();
    __syncthreads();
    if (kb + 32 < FEATS)
      async_b128(cur ? bdst0 : bdst1, bsrc + kb + 32);
    v16h a_next = a_cur;
    if (kb + 32 < FEATS) a_next = load_a(Abase + kb + 32, FEATS);
    const _Float16* bb = &bbuf[cur][0];
    v16h bf[4];
#pragma unroll
    for (int t = 0; t < 4; ++t)
      bf[t] = load_b(bb + t * 16 * BPAD, BPAD);
#pragma unroll
    for (int t = 0; t < 4; ++t)
      acc[t] = wmma_f16(a_cur, bf[t], acc[t]);
    a_cur = a_next;
    cur ^= 1;
  }

  const int lane = tid & 31;
  const int nloc = lane & 15, hb = lane >> 4;
#pragma unroll
  for (int t = 0; t < 4; ++t) {
    const int col = cg * 64 + t * 16 + nloc;
    const float bc = bo[col];
#pragma unroll
    for (int v = 0; v < 8; ++v) {
      const int row = rowtile * 16 + v + 8 * hb;
      out[(size_t)row * FEATS + col] = acc[t][v] + bc;
    }
  }
}

// ---- launcher ---------------------------------------------------------------

extern "C" void kernel_launch(void* const* d_in, const int* in_sizes, int n_in,
                              void* d_out, int out_size, void* d_ws, size_t ws_size,
                              hipStream_t stream) {
  const float* x  = (const float*)d_in[0];
  const float* Wq = (const float*)d_in[1];
  const float* bq = (const float*)d_in[2];
  const float* Wk = (const float*)d_in[3];
  const float* bk = (const float*)d_in[4];
  const float* Wv = (const float*)d_in[5];
  const float* bv = (const float*)d_in[6];
  const float* Wo = (const float*)d_in[7];
  const float* bo = (const float*)d_in[8];
  float* out = (float*)d_out;

  _Float16* ws = (_Float16*)d_ws;
  const size_t NX = (size_t)NTOK * FEATS;    // 4M halfs
  const size_t NW = (size_t)FEATS * FEATS;   // 1M halfs
  _Float16* xh  = ws;
  _Float16* Wqh = xh  + NX;
  _Float16* Wkh = Wqh + NW;
  _Float16* Wvh = Wkh + NW;
  _Float16* Wvx = Wvh;  // alias clarity only
  _Float16* Woh = Wvh + NW;
  _Float16* Qh  = Woh + NW;      // [B,H,N,DH]
  _Float16* Kh  = Qh  + NX;      // [B,H,N,DH]
  _Float16* Vt  = Kh  + NX;      // [B,H,DH,N]
  _Float16* AO  = Vt  + NX;      // [B,N,F]
  (void)Wvx;
  // workspace: 24M halfs = 48 MB

  cvt_f32_f16<<<2048, 256, 0, stream>>>(x,  xh,  (int)NX);
  cvt_f32_f16<<<1024, 256, 0, stream>>>(Wq, Wqh, (int)NW);
  cvt_f32_f16<<<1024, 256, 0, stream>>>(Wk, Wkh, (int)NW);
  cvt_f32_f16<<<1024, 256, 0, stream>>>(Wv, Wvh, (int)NW);
  cvt_f32_f16<<<1024, 256, 0, stream>>>(Wo, Woh, (int)NW);

  qkv_gemm<<<512, 256, 0, stream>>>(xh, Wqh, bq, Qh, 0);
  qkv_gemm<<<512, 256, 0, stream>>>(xh, Wkh, bk, Kh, 1);
  qkv_gemm<<<512, 256, 0, stream>>>(xh, Wvh, bv, Vt, 2);

  flash_attn<<<512, 256, 0, stream>>>(Qh, Kh, Vt, AO);

  out_gemm<<<512, 256, 0, stream>>>(AO, Woh, bo, out);
}
